// AddDotProductAttention_15685220565607
// MI455X (gfx1250) — matrix operand order repaired and checked by reading, not compiled, not run on hardware
//
#include <hip/hip_runtime.h>
#include <hip/hip_bf16.h>

#define BB 4
#define HHN 8
#define SS 1024
#define DDIM 64
#define KTILE 64
#define NKT (SS / KTILE)   // 16 k-tiles of 64

typedef __attribute__((ext_vector_type(16))) __bf16 v16bf;
typedef __attribute__((ext_vector_type(8)))  float  v8f;

union FragBF {
  v16bf v;
  unsigned int u[8];
  __bf16 e[16];
};
union FragF {
  v8f v;
  float e[8];
};

// ---- pre-pass: fp32 -> bf16 (optionally scaled) ----
__global__ void cvt_scale_bf16(const float* __restrict__ in, __bf16* __restrict__ out,
                               float scale, int n) {
  int i = blockIdx.x * blockDim.x + threadIdx.x;
  if (i < n) out[i] = (__bf16)(in[i] * scale);
}

// ---- pre-pass: v [B,H,S,D] fp32 -> vT [B,H,D,S] bf16 ----
__global__ void cvt_transpose_v(const float* __restrict__ v, __bf16* __restrict__ vT) {
  int i = blockIdx.x * blockDim.x + threadIdx.x;   // over B*H*D*S outputs
  int s  = i & (SS - 1);
  int d  = (i >> 10) & (DDIM - 1);
  int bh = i >> 16;
  vT[i] = (__bf16)v[((size_t)bh * SS + s) * DDIM + d];
}

// ---- fused talking-heads attention, one (b, 16-query tile) per block ----
// LDS (dynamic, ~290 KB -- uses CDNA5's 320 KB WGP LDS):
//   [0, 256K)          : f16 cache of MIXED scores s'[g][kt][t][j][lane]
//                        (wave-private per g: written+read by wave g only)
//   [256K, 256K+34K)   : pass 1: raw score sharing lds_s[h][t][j][lane] (32 KB)
//                        pass 2: probabilities lds_p[g][kk][q pad17]     (34 KB)
//   tail               : biasmix[g][q] (512 B)
__global__ __launch_bounds__(256)
void talking_heads_attn(const __bf16* __restrict__ qbf,
                        const __bf16* __restrict__ kbf,
                        const __bf16* __restrict__ vTbf,
                        const float* __restrict__ attn_bias,
                        const float* __restrict__ Wl, const float* __restrict__ bl,
                        const float* __restrict__ Ww, const float* __restrict__ bw,
                        float* __restrict__ out) {
  extern __shared__ char smem[];
  _Float16* lds_sc = (_Float16*)smem;                    // 8*16*4*8*32 f16 = 256 KB
  float*    lds_s  = (float*)(smem + 262144);            // pass-1 raw scores (32 KB)
  float*    lds_p  = (float*)(smem + 262144);            // pass-2 probs (aliases lds_s)
  float*    lds_bm = (float*)(smem + 262144 + 34816);    // biasmix (512 B)

  const int w    = threadIdx.x >> 5;    // wave id == head role (h producer, g consumer)
  const int lane = threadIdx.x & 31;
  const int half = lane >> 4;
  const int lq   = lane & 15;
  const int q0   = blockIdx.x * 16;
  const int b    = blockIdx.y;

  // small talking-heads weights, per wave
  float wl[HHN], ww[HHN];
#pragma unroll
  for (int h = 0; h < HHN; ++h) wl[h] = Wl[w * HHN + h];
#pragma unroll
  for (int g = 0; g < HHN; ++g) ww[g] = Ww[w * HHN + g];
  const float blw = bl[w];
  const float bww = bw[w];

  // biasmix[g][q] = sum_h Wl[g,h] * attn_bias[b,h,q0+q]   (bl folded in)
  if (lane < 16) {
    float bm = blw;
#pragma unroll
    for (int h = 0; h < HHN; ++h)
      bm += wl[h] * attn_bias[((size_t)b * HHN + h) * SS + q0 + lane];
    lds_bm[w * 16 + lane] = bm;
  }

  // Q A-fragments (16x32 bf16 A layout), head w, softmax scale folded into qbf
  FragBF qa[2];
  {
    const unsigned int* qrow =
        (const unsigned int*)(qbf + (((size_t)(b * HHN + w)) * SS + q0 + lq) * DDIM);
#pragma unroll
    for (int c = 0; c < 2; ++c)
#pragma unroll
      for (int j = 0; j < 8; ++j) {
        int dbase = c * 32 + 2 * (j & 3) + 16 * (j >> 2) + 8 * half;
        qa[c].u[j] = qrow[dbase >> 1];
      }
  }

  __syncthreads();

  float m[8], l[8];
#pragma unroll
  for (int j = 0; j < 8; ++j) { m[j] = -3.0e38f; l[j] = 0.f; }

  const __bf16* kbase  = kbf  + ((size_t)(b * HHN + w)) * SS * DDIM;
  const __bf16* vTbase = vTbf + ((size_t)(b * HHN + w)) * DDIM * SS;

  // ========== pass 1: QK^T once, cross-head mix once, cache s', online stats ==========
  for (int kt = 0; kt < NKT; ++kt) {
    if (kt + 1 < NKT)
      __builtin_prefetch(kbase + ((size_t)((kt + 1) * KTILE + lq)) * DDIM, 0, 1);
#pragma unroll
    for (int t = 0; t < 4; ++t) {
      FragF acc;
#pragma unroll
      for (int i = 0; i < 8; ++i) acc.e[i] = 0.f;
      const unsigned int* krow =
          (const unsigned int*)(kbase + ((size_t)(kt * KTILE + t * 16 + lq)) * DDIM);
#pragma unroll
      for (int c = 0; c < 2; ++c) {
        FragBF kb;
#pragma unroll
        for (int j = 0; j < 8; ++j)
          kb.u[j] = krow[(c * 32 + 2 * (j & 3) + 16 * (j >> 2) + 8 * half) >> 1];
        acc.v = __builtin_amdgcn_wmma_f32_16x16x32_bf16(
            false, qa[c].v, false, kb.v, (short)0, acc.v, false, false);
      }
#pragma unroll
      for (int j = 0; j < 8; ++j)
        lds_s[((w * 4 + t) * 8 + j) * 32 + lane] = acc.e[j];
    }
    __syncthreads();

    float sp[4][8];
#pragma unroll
    for (int t = 0; t < 4; ++t)
#pragma unroll
      for (int j = 0; j < 8; ++j) {
        float s = lds_bm[w * 16 + j + 8 * half];
#pragma unroll
        for (int h = 0; h < HHN; ++h)
          s += wl[h] * lds_s[((h * 4 + t) * 8 + j) * 32 + lane];
        sp[t][j] = s;
        // cache mixed score (wave-private region: no barrier needed for this buffer)
        lds_sc[(((w * NKT + kt) * 4 + t) * 8 + j) * 32 + lane] = (_Float16)s;
      }
#pragma unroll
    for (int j = 0; j < 8; ++j) {
      float rmax = sp[0][j];
#pragma unroll
      for (int t = 1; t < 4; ++t) rmax = fmaxf(rmax, sp[t][j]);
#pragma unroll
      for (int off = 1; off < 16; off <<= 1)
        rmax = fmaxf(rmax, __shfl_xor(rmax, off, 32));
      float mn = fmaxf(m[j], rmax);
      float rs = 0.f;
#pragma unroll
      for (int t = 0; t < 4; ++t) rs += __expf(sp[t][j] - mn);
#pragma unroll
      for (int off = 1; off < 16; off <<= 1)
        rs += __shfl_xor(rs, off, 32);
      l[j] = l[j] * __expf(m[j] - mn) + rs;
      m[j] = mn;
    }
    __syncthreads();   // protect lds_s before next tile overwrites it
  }

  float inv_l[8];
#pragma unroll
  for (int j = 0; j < 8; ++j) inv_l[j] = 1.0f / l[j];

  // ========== pass 2: p from cached s', post-softmax mix, P@V ==========
  FragF o[4];
#pragma unroll
  for (int ds = 0; ds < 4; ++ds)
#pragma unroll
    for (int i = 0; i < 8; ++i) o[ds].e[i] = 0.f;

  for (int kt = 0; kt < NKT; ++kt) {
    // normalized probabilities for mixed head w -> lds_p[g][kk][q]
#pragma unroll
    for (int t = 0; t < 4; ++t)
#pragma unroll
      for (int j = 0; j < 8; ++j) {
        float s = (float)lds_sc[(((w * NKT + kt) * 4 + t) * 8 + j) * 32 + lane];
        float p = __expf(s - m[j]) * inv_l[j];
        lds_p[(w * 64 + t * 16 + lq) * 17 + j + 8 * half] = p;
      }
    __syncthreads();

    // w_h = bw + sum_g Ww[h,g] p_g, then O_h += W @ V_h with WMMA
#pragma unroll
    for (int c = 0; c < 2; ++c) {
      FragBF wa;
#pragma unroll
      for (int j = 0; j < 8; ++j) {
        int kk = c * 32 + 2 * (j & 3) + 16 * (j >> 2) + 8 * half;
        float w0 = bww, w1 = bww;
#pragma unroll
        for (int g = 0; g < HHN; ++g) {
          w0 += ww[g] * lds_p[(g * 64 + kk) * 17 + lq];
          w1 += ww[g] * lds_p[(g * 64 + kk + 1) * 17 + lq];
        }
        wa.e[2 * j]     = (__bf16)w0;
        wa.e[2 * j + 1] = (__bf16)w1;
      }
#pragma unroll
      for (int ds = 0; ds < 4; ++ds) {
        FragBF vb;
        const unsigned int* vrow =
            (const unsigned int*)(vTbase + ((size_t)(ds * 16 + lq)) * SS +
                                  kt * KTILE + c * 32);
#pragma unroll
        for (int j = 0; j < 8; ++j) vb.u[j] = vrow[(2 * (j & 3) + 16 * (j >> 2) + 8 * half) >> 1];
        o[ds].v = __builtin_amdgcn_wmma_f32_16x16x32_bf16(
            false, wa.v, false, vb.v, (short)0, o[ds].v, false, false);
      }
    }
    __syncthreads();   // protect lds_p before next tile overwrites it
  }

  // write output [B,H,S,D] fp32
#pragma unroll
  for (int ds = 0; ds < 4; ++ds)
#pragma unroll
    for (int j = 0; j < 8; ++j)
      out[(((size_t)(b * HHN + w)) * SS + q0 + j + 8 * half) * DDIM + ds * 16 + lq] =
          o[ds].e[j];
}

extern "C" void kernel_launch(void* const* d_in, const int* in_sizes, int n_in,
                              void* d_out, int out_size, void* d_ws, size_t ws_size,
                              hipStream_t stream) {
  (void)in_sizes; (void)n_in; (void)out_size; (void)ws_size;
  const float* q         = (const float*)d_in[0];
  const float* k         = (const float*)d_in[1];
  const float* v         = (const float*)d_in[2];
  const float* attn_bias = (const float*)d_in[3];
  const float* Wl        = (const float*)d_in[4];
  const float* bl        = (const float*)d_in[5];
  const float* Ww        = (const float*)d_in[6];
  const float* bw        = (const float*)d_in[7];
  float* out = (float*)d_out;

  const int N = BB * HHN * SS * DDIM;  // 2,097,152
  __bf16* qbf = (__bf16*)d_ws;
  __bf16* kbf = qbf + N;
  __bf16* vT  = kbf + N;

  dim3 cgrid((N + 255) / 256), cblk(256);
  cvt_scale_bf16<<<cgrid, cblk, 0, stream>>>(q, qbf, 0.125f, N);  // 1/sqrt(64)
  cvt_scale_bf16<<<cgrid, cblk, 0, stream>>>(k, kbf, 1.0f, N);
  cvt_transpose_v<<<cgrid, cblk, 0, stream>>>(v, vT);

  dim3 grid(SS / 16, BB), blk(256);
  size_t shmem = 262144 + 34816 + 512;  // ~290 KB of the 320 KB WGP LDS
  talking_heads_attn<<<grid, blk, shmem, stream>>>(qbf, kbf, vT, attn_bias,
                                                   Wl, bl, Ww, bw, out);
}
